// EncoderLayer_28406913696385
// MI455X (gfx1250) — matrix operand order repaired and checked by reading, not compiled, not run on hardware
//
#include <hip/hip_runtime.h>

// ---------------------------------------------------------------------------
// MI455X (gfx1250, wave32) transformer encoder layer.
// All GEMMs + attention run on v_wmma_f32_16x16x32_f16 (f16 in, f32 acc).
// Weights/V are pre-packed into exact WMMA B-fragment layout so the inner
// loops are pure b128 loads + WMMA. Attention is a register-resident
// flash-attention computing S^T = K*Q^T so softmax state stays in-lane.
// GEMM waves compute 32x64 output tiles (2 A-frags x 4 B-frags per k-step)
// and prefetch the next k-step's packed-B line into the WGP cache.
// ---------------------------------------------------------------------------

typedef _Float16 f16t;
typedef __attribute__((ext_vector_type(16))) _Float16 v16h;
typedef __attribute__((ext_vector_type(8)))  _Float16 v8h;
typedef __attribute__((ext_vector_type(8)))  float    v8f;

__device__ __forceinline__ v8f wmma_f16f32(v16h a, v16h b, v8f c) {
  // (neg_a, A, neg_b, B, c_mod, C, reuse_a, reuse_b)
  return __builtin_amdgcn_wmma_f32_16x16x32_f16(false, a, false, b, (short)0, c,
                                                false, false);
}

// A-fragment (16x32 f16, row-major source): lane m (0-15) holds row m,
// elements 0..7 = K[base..base+7], 8..15 = K[base+16..base+23];
// lanes 16-31 shift the k-base by +8. -> two 16B loads per lane.
__device__ __forceinline__ v16h load_a_pair(const f16t* __restrict__ p) {
  v8h lo = *(const v8h*)p;
  v8h hh = *(const v8h*)(p + 16);
  v16h r;
#pragma unroll
  for (int i = 0; i < 8; ++i) { r[i] = lo[i]; r[i + 8] = hh[i]; }
  return r;
}

// ---------------------------------------------------------------------------
// Elementwise convert f32 -> f16
// ---------------------------------------------------------------------------
__global__ void cvt_f32_f16(const float* __restrict__ src,
                            f16t* __restrict__ dst, int n) {
  int i = blockIdx.x * 256 + threadIdx.x;
  if (i < n) dst[i] = (f16t)src[i];
}

// ---------------------------------------------------------------------------
// Pack a row-major f32 [K x N] matrix into WMMA B-fragment order:
// frag(kt,nt)[lane][h] = B[kt*32 + 16*(h>=8) + 8*(lane>=16) + (h&7)][nt*16 + (lane&15)]
// ---------------------------------------------------------------------------
__global__ void pack_b_f32(const float* __restrict__ src,
                           f16t* __restrict__ dst, int K, int N) {
  size_t t = (size_t)blockIdx.x * 256 + threadIdx.x;
  if (t >= (size_t)K * N) return;
  int h    = (int)(t & 15);
  int lane = (int)((t >> 4) & 31);
  size_t f = t >> 9;
  int nt16 = N >> 4;
  int kt = (int)(f / nt16), nt = (int)(f % nt16);
  int row = kt * 32 + ((h >> 3) << 4) + ((lane >> 4) << 3) + (h & 7);
  int col = nt * 16 + (lane & 15);
  dst[t] = (f16t)src[(size_t)row * N + col];
}

// Pack V (f16, 32 contiguous head-slabs of [2048 x 64]) into B-fragments.
__global__ void pack_v_f16(const f16t* __restrict__ src,
                           f16t* __restrict__ dst) {
  size_t t = (size_t)blockIdx.x * 256 + threadIdx.x;   // 32*2048*64 total
  size_t slab = t >> 17;            // 131072 halves per (b,h) slab
  size_t u = t & 131071;
  int h    = (int)(u & 15);
  int lane = (int)((u >> 4) & 31);
  int f    = (int)(u >> 9);         // 64 k-tiles x 4 d-tiles
  int kt = f >> 2, nt = f & 3;
  int row = kt * 32 + ((h >> 3) << 4) + ((lane >> 4) << 3) + (h & 7);
  int col = nt * 16 + (lane & 15);
  dst[t] = src[slab * 131072 + (size_t)row * 64 + col];
}

// ---------------------------------------------------------------------------
// GEMM: out[M x N] = A[M x K](f16 row-major) @ Bpacked + bias (+res) (ReLU?)
// One wave computes a 32x64 tile: 2 A-frags x 4 B-frags -> 8 WMMAs / k-step.
// ---------------------------------------------------------------------------
template <int OUT_F16, int RELU>
__global__ void gemm_kernel(const f16t* __restrict__ A,
                            const f16t* __restrict__ Bp,
                            const float* __restrict__ bias,
                            const float* __restrict__ res,
                            float* __restrict__ outF,
                            f16t* __restrict__ outH,
                            int M, int N, int K) {
  int lane = threadIdx.x & 31;
  int wid  = blockIdx.x * (blockDim.x >> 5) + (threadIdx.x >> 5);
  int nblk = N >> 6;
  int mt = wid / nblk, nb = wid % nblk;
  if (mt * 32 >= M) return;                 // wave-uniform
  int m0 = mt * 32, n0 = nb * 64;
  int hi = lane >> 4;
  int row0 = m0 + (lane & 15);
  int row1 = row0 + 16;
  int nt16 = N >> 4;
  const size_t bstride = (size_t)nt16 * 512;   // packed halves per k-tile

  v8f c[2][4];
#pragma unroll
  for (int i = 0; i < 2; ++i)
#pragma unroll
    for (int j = 0; j < 4; ++j) c[i][j] = (v8f){};

  for (int k0 = 0; k0 < K; k0 += 32) {
    v16h a0 = load_a_pair(A + (size_t)row0 * K + k0 + hi * 8);
    v16h a1 = load_a_pair(A + (size_t)row1 * K + k0 + hi * 8);
    const f16t* bb =
        Bp + (size_t)(k0 >> 5) * bstride + (size_t)(n0 >> 4) * 512 + lane * 16;
    if (k0 + 32 < K)                       // pull next k-step's B line to WGP$
      __builtin_prefetch(bb + bstride, 0, 3);
    v16h b0 = *(const v16h*)(bb);
    v16h b1 = *(const v16h*)(bb + 512);
    v16h b2 = *(const v16h*)(bb + 1024);
    v16h b3 = *(const v16h*)(bb + 1536);
    c[0][0] = wmma_f16f32(a0, b0, c[0][0]);
    c[1][0] = wmma_f16f32(a1, b0, c[1][0]);
    c[0][1] = wmma_f16f32(a0, b1, c[0][1]);
    c[1][1] = wmma_f16f32(a1, b1, c[1][1]);
    c[0][2] = wmma_f16f32(a0, b2, c[0][2]);
    c[1][2] = wmma_f16f32(a1, b2, c[1][2]);
    c[0][3] = wmma_f16f32(a0, b3, c[0][3]);
    c[1][3] = wmma_f16f32(a1, b3, c[1][3]);
  }

#pragma unroll
  for (int i = 0; i < 2; ++i) {
#pragma unroll
    for (int j = 0; j < 4; ++j) {
      int col = n0 + j * 16 + (lane & 15);
      float bcol = bias[col];
#pragma unroll
      for (int r = 0; r < 8; ++r) {
        int rr = m0 + i * 16 + r + hi * 8;
        float v = c[i][j][r] + bcol;
        if (res) v += res[(size_t)rr * N + col];
        if (RELU) v = v > 0.f ? v : 0.f;
        if (OUT_F16) outH[(size_t)rr * N + col] = (f16t)v;
        else         outF[(size_t)rr * N + col] = v;
      }
    }
  }
}

// ---------------------------------------------------------------------------
// Fused flash attention. Per (b,h): contiguous [2048 x 64] Q/K/V slabs
// (the reference's .view() split is a pure reinterpretation).
// Each wave owns 16 query rows. Computes S^T = K*Q^T so per-lane softmax
// state is for query (lane&15); online max/sum merged with shfl_xor(16).
// ---------------------------------------------------------------------------
__global__ void attn_kernel(const f16t* __restrict__ Q,
                            const f16t* __restrict__ Km,
                            const f16t* __restrict__ Vp,
                            const int* __restrict__ mask,
                            f16t* __restrict__ O) {
  const int L = 2048, DT = 64;
  int lane = threadIdx.x & 31;
  int wib  = threadIdx.x >> 5;           // 4 waves / block
  int bh   = blockIdx.x >> 5;            // 32 blocks per (b,h)
  int qb   = blockIdx.x & 31;
  int q0   = (qb * 4 + wib) * 16;
  int hi   = lane >> 4;
  int qi   = lane & 15;
  size_t slab = (size_t)bh * (L * DT);
  const f16t* Qb = Q + slab;
  const f16t* Kb = Km + slab;
  const f16t* Vb = Vp + slab;
  const int* mb = mask + (bh >> 4) * L;  // batch = bh/16

  // Q^T B-fragments (held in regs for whole k loop): lane n gets row q0+n,
  // d = 32s + 8*hi + [0,8) and 32s + 16 + 8*hi + [0,8).
  v16h bq0 = load_a_pair(Qb + (size_t)(q0 + qi) * DT + 0  + hi * 8);
  v16h bq1 = load_a_pair(Qb + (size_t)(q0 + qi) * DT + 32 + hi * 8);

  v8f o[4];
#pragma unroll
  for (int j = 0; j < 4; ++j) o[j] = (v8f){};
  float mrun = -1e30f, lrun = 0.f;

  for (int kb = 0; kb < L; kb += 32) {
    // K rows as A-matrix (two 16-row sub-tiles, two d-steps each)
    const f16t* kr0 = Kb + (size_t)(kb + qi) * DT + hi * 8;
    const f16t* kr1 = Kb + (size_t)(kb + 16 + qi) * DT + hi * 8;
    v16h a00 = load_a_pair(kr0);
    v16h a01 = load_a_pair(kr0 + 32);
    v16h a10 = load_a_pair(kr1);
    v16h a11 = load_a_pair(kr1 + 32);
    v8f st0 = (v8f){}, st1 = (v8f){};
    st0 = wmma_f16f32(a00, bq0, st0);
    st0 = wmma_f16f32(a01, bq1, st0);
    st1 = wmma_f16f32(a10, bq0, st1);
    st1 = wmma_f16f32(a11, bq1, st1);

    // scale 1/sqrt(64) + faithful mask (==0 -> -1e-12)
    const int* mq0 = mb + kb + hi * 8;
    const int* mq1 = mb + kb + 16 + hi * 8;
    float s0[8], s1[8];
#pragma unroll
    for (int r = 0; r < 8; ++r) {
      float a = st0[r] * 0.125f, b = st1[r] * 0.125f;
      s0[r] = (mq0[r] == 0) ? -1e-12f : a;
      s1[r] = (mq1[r] == 0) ? -1e-12f : b;
    }

    float tmax = s0[0];
#pragma unroll
    for (int r = 0; r < 8; ++r) {
      tmax = fmaxf(tmax, s0[r]); tmax = fmaxf(tmax, s1[r]);
    }
    tmax = fmaxf(tmax, __shfl_xor(tmax, 16, 32));
    float mnew  = fmaxf(mrun, tmax);
    float alpha = __expf(mrun - mnew);

    float p0[8], p1[8], rsum = 0.f;
#pragma unroll
    for (int r = 0; r < 8; ++r) {
      p0[r] = __expf(s0[r] - mnew);
      p1[r] = __expf(s1[r] - mnew);
      rsum += p0[r] + p1[r];
    }
    rsum += __shfl_xor(rsum, 16, 32);
    lrun = lrun * alpha + rsum;
    mrun = mnew;

    // rescale O accumulators: element r belongs to query row (r + 8*hi),
    // whose alpha lives in lane (r + 8*hi)
    float af[8];
#pragma unroll
    for (int r = 0; r < 8; ++r) af[r] = __shfl(alpha, r + 8 * hi, 32);
#pragma unroll
    for (int j = 0; j < 4; ++j)
#pragma unroll
      for (int r = 0; r < 8; ++r) o[j][r] *= af[r];

    // P A-fragment: pure in-lane repack of the S^T accumulators
    v16h pa;
#pragma unroll
    for (int r = 0; r < 8; ++r) {
      pa[r]     = (f16t)p0[r];
      pa[r + 8] = (f16t)p1[r];
    }

    const f16t* vb = Vb + (size_t)(kb >> 5) * 4 * 512 + lane * 16;
    v16h v0 = *(const v16h*)(vb);
    v16h v1 = *(const v16h*)(vb + 512);
    v16h v2 = *(const v16h*)(vb + 1024);
    v16h v3 = *(const v16h*)(vb + 1536);
    o[0] = wmma_f16f32(pa, v0, o[0]);
    o[1] = wmma_f16f32(pa, v1, o[1]);
    o[2] = wmma_f16f32(pa, v2, o[2]);
    o[3] = wmma_f16f32(pa, v3, o[3]);
  }

  float inv = 1.f / lrun;
  float ir[8];
#pragma unroll
  for (int r = 0; r < 8; ++r) ir[r] = __shfl(inv, r + 8 * hi, 32);
#pragma unroll
  for (int j = 0; j < 4; ++j)
#pragma unroll
    for (int r = 0; r < 8; ++r)
      O[slab + (size_t)(q0 + r + 8 * hi) * DT + j * 16 + qi] =
          (f16t)(o[j][r] * ir[r]);
}

// ---------------------------------------------------------------------------
// LayerNorm, faithful: mean, unbiased std (ddof=1), y = g*(x-m)/(std+eps)+b
// One block (8 waves) per row of 1024.
// ---------------------------------------------------------------------------
__global__ void ln_kernel(const float* __restrict__ in,
                          const float* __restrict__ gamma,
                          const float* __restrict__ beta,
                          float* __restrict__ outF,
                          f16t* __restrict__ outH) {
  const int D = 1024;
  const float* p = in + (size_t)blockIdx.x * D;
  float s = 0.f, s2 = 0.f;
  for (int i = threadIdx.x; i < D; i += 256) {
    float v = p[i]; s += v; s2 += v * v;
  }
#pragma unroll
  for (int off = 16; off; off >>= 1) {
    s  += __shfl_down(s, off, 32);
    s2 += __shfl_down(s2, off, 32);
  }
  __shared__ float rs[8], rq[8];
  int w = threadIdx.x >> 5, l = threadIdx.x & 31;
  if (l == 0) { rs[w] = s; rq[w] = s2; }
  __syncthreads();
  if (threadIdx.x == 0) {
    float a = 0.f, b = 0.f;
    for (int i = 0; i < 8; ++i) { a += rs[i]; b += rq[i]; }
    rs[0] = a; rq[0] = b;
  }
  __syncthreads();
  float mean = rs[0] * (1.f / 1024.f);
  float var  = (rq[0] - 1024.f * mean * mean) * (1.f / 1023.f);
  var = var > 0.f ? var : 0.f;
  float inv = 1.f / (sqrtf(var) + 1e-12f);
  for (int i = threadIdx.x; i < D; i += 256) {
    float y = gamma[i] * ((p[i] - mean) * inv) + beta[i];
    if (outF) outF[(size_t)blockIdx.x * D + i] = y;
    if (outH) outH[(size_t)blockIdx.x * D + i] = (f16t)y;
  }
}

// ---------------------------------------------------------------------------
extern "C" void kernel_launch(void* const* d_in, const int* in_sizes, int n_in,
                              void* d_out, int out_size, void* d_ws,
                              size_t ws_size, hipStream_t stream) {
  (void)in_sizes; (void)n_in; (void)out_size; (void)ws_size;
  const float* x     = (const float*)d_in[0];
  const int*   mask  = (const int*)d_in[1];
  const float* wq_w  = (const float*)d_in[2];
  const float* wq_b  = (const float*)d_in[3];
  const float* wk_w  = (const float*)d_in[4];
  const float* wk_b  = (const float*)d_in[5];
  const float* wv_w  = (const float*)d_in[6];
  const float* wv_b  = (const float*)d_in[7];
  const float* wo_w  = (const float*)d_in[8];
  const float* wo_b  = (const float*)d_in[9];
  const float* g1    = (const float*)d_in[10];
  const float* b1    = (const float*)d_in[11];
  const float* f1_w  = (const float*)d_in[12];
  const float* f1_b  = (const float*)d_in[13];
  const float* f2_w  = (const float*)d_in[14];
  const float* f2_b  = (const float*)d_in[15];
  const float* g2    = (const float*)d_in[16];
  const float* b2    = (const float*)d_in[17];

  char* ws = (char*)d_ws;
  const size_t MB = (size_t)1 << 20;
  f16t*  xh  = (f16t*)(ws + 0);        //  8 MB  x in f16
  f16t*  wqp = (f16t*)(ws + 8 * MB);   //  2 MB  packed weights
  f16t*  wkp = (f16t*)(ws + 10 * MB);
  f16t*  wvp = (f16t*)(ws + 12 * MB);
  f16t*  wop = (f16t*)(ws + 14 * MB);
  f16t*  f1p = (f16t*)(ws + 16 * MB);  //  8 MB
  f16t*  f2p = (f16t*)(ws + 24 * MB);  //  8 MB
  f16t*  q   = (f16t*)(ws + 32 * MB);  //  8 MB
  f16t*  kk  = (f16t*)(ws + 40 * MB);  //  8 MB
  f16t*  v   = (f16t*)(ws + 48 * MB);  //  8 MB
  f16t*  vp  = (f16t*)(ws + 56 * MB);  //  8 MB packed V frags
  f16t*  o   = (f16t*)(ws + 64 * MB);  //  8 MB attn out
  float* s1  = (float*)(ws + 72 * MB); // 16 MB (reused: attn+res, ffn2+res)
  float* x1  = (float*)(ws + 88 * MB); // 16 MB
  f16t*  x1h = (f16t*)(ws + 104 * MB); //  8 MB
  f16t*  h1  = (f16t*)(ws + 112 * MB); // 32 MB  (total 144 MB)

  cvt_f32_f16<<<16384, 256, 0, stream>>>(x, xh, 4096 * 1024);
  pack_b_f32<<<4096, 256, 0, stream>>>(wq_w, wqp, 1024, 1024);
  pack_b_f32<<<4096, 256, 0, stream>>>(wk_w, wkp, 1024, 1024);
  pack_b_f32<<<4096, 256, 0, stream>>>(wv_w, wvp, 1024, 1024);
  pack_b_f32<<<4096, 256, 0, stream>>>(wo_w, wop, 1024, 1024);
  pack_b_f32<<<16384, 256, 0, stream>>>(f1_w, f1p, 1024, 4096);
  pack_b_f32<<<16384, 256, 0, stream>>>(f2_w, f2p, 4096, 1024);

  // Q, K, V projections (f16 out): (4096/32)*(1024/64) = 2048 waves
  gemm_kernel<1, 0><<<512, 128, 0, stream>>>(xh, wqp, wq_b, nullptr,
                                             nullptr, q, 4096, 1024, 1024);
  gemm_kernel<1, 0><<<512, 128, 0, stream>>>(xh, wkp, wk_b, nullptr,
                                             nullptr, kk, 4096, 1024, 1024);
  gemm_kernel<1, 0><<<512, 128, 0, stream>>>(xh, wvp, wv_b, nullptr,
                                             nullptr, v, 4096, 1024, 1024);
  pack_v_f16<<<16384, 256, 0, stream>>>(v, vp);

  // Fused flash attention -> o (f16, contiguous head slabs = [B,L,D] view)
  attn_kernel<<<1024, 128, 0, stream>>>(q, kk, vp, mask, o);

  // out-proj + residual x -> s1 (f32), then LN1 -> x1 (f32) + x1h (f16)
  gemm_kernel<0, 0><<<512, 128, 0, stream>>>(o, wop, wo_b, x,
                                             s1, nullptr, 4096, 1024, 1024);
  ln_kernel<<<4096, 256, 0, stream>>>(s1, g1, b1, x1, x1h);

  // FFN: (4096/32)*(4096/64) = 8192 waves ; then (4096/32)*(1024/64) = 2048
  gemm_kernel<1, 1><<<2048, 128, 0, stream>>>(x1h, f1p, f1_b, nullptr,
                                              nullptr, h1, 4096, 4096, 1024);
  gemm_kernel<0, 0><<<512, 128, 0, stream>>>(h1, f2p, f2_b, x1,
                                             s1, nullptr, 4096, 1024, 4096);
  ln_kernel<<<4096, 256, 0, stream>>>(s1, g2, b2, (float*)d_out, nullptr);
}
